// Seq2Seq_74947179315483
// MI455X (gfx1250) — matrix operand order repaired and checked by reading, not compiled, not run on hardware
//
#include <hip/hip_runtime.h>
#include <hip/hip_bf16.h>
#include <math.h>

// ---------------- model dimensions ----------------
#define BB    32
#define TT    1664
#define FIN   80
#define FPAD  96       // FIN padded to multiple of 32
#define HENC  256      // encoder hidden per direction, gates = 1024
#define KVD   128
#define DDEC  512      // decoder1 hidden, gates = 2048
#define EMBD  256
#define VOC   30
#define LDECS 200
#define SENC  208      // 1664 / 8
#define TCH   104      // time chunk (divides 1664, 832, 416, 208)
#define PRED  (BB * LDECS * VOC)   // 192000, offset of attn0 in d_out

typedef __bf16 bf16_t;
typedef __attribute__((ext_vector_type(16))) bf16_t v16bf;
typedef __attribute__((ext_vector_type(8)))  bf16_t v8bf;
typedef __attribute__((ext_vector_type(8)))  float  v8f;

static __device__ __forceinline__ unsigned short f2bf(float f) {
  unsigned u = __builtin_bit_cast(unsigned, f);
  u += 0x7FFFu + ((u >> 16) & 1u);            // round-to-nearest-even
  return (unsigned short)(u >> 16);
}
static __device__ __forceinline__ float bf2f(unsigned short u) {
  unsigned v = ((unsigned)u) << 16;
  return __builtin_bit_cast(float, v);
}
static __device__ __forceinline__ float sigm(float x) { return 1.0f / (1.0f + __expf(-x)); }

// contiguous 16 bf16 -> B fragment (K = j + 16*grp is consecutive memory)
static __device__ __forceinline__ v16bf loadB16(const unsigned short* p) {
  return *(const v16bf*)p;
}
// contiguous 16 bf16 -> A fragment (K = j + 16*grp is consecutive memory)
static __device__ __forceinline__ v16bf loadA16(const unsigned short* p) {
  return *(const v16bf*)p;
}

// ---------------- conversion helpers ----------------
__global__ void k_cvt(const float* __restrict__ s, unsigned short* __restrict__ d, int n) {
  for (int i = blockIdx.x * blockDim.x + threadIdx.x; i < n; i += gridDim.x * blockDim.x)
    d[i] = f2bf(s[i]);
}
// fp32 [rows,Ks] -> bf16 [rows,Kd] zero-padded
__global__ void k_cvt_pad(const float* __restrict__ s, unsigned short* __restrict__ d,
                          int rows, int Ks, int Kd) {
  const int n = rows * Kd;
  for (int i = blockIdx.x * blockDim.x + threadIdx.x; i < n; i += gridDim.x * blockDim.x) {
    const int r = i / Kd, k = i - r * Kd;
    d[i] = (k < Ks) ? f2bf(s[r * Ks + k]) : (unsigned short)0;
  }
}
__global__ void k_zero_u32(unsigned* __restrict__ p, int n) {
  const int i = blockIdx.x * blockDim.x + threadIdx.x;
  if (i < n) p[i] = 0u;
}

// ---------------- generic Y[m,n] = A_rows @ W^T + bias (bf16 WMMA, fp32 acc) ----
// K must be a multiple of 32. A is bf16; rows: m -> (b = m/Tc, t = t0 + m%Tc),
// row ptr = A + (b*Ttot + t)*K. W is bf16 [N,K] row-major. Y: [32*Tc, N] fp32.
__global__ void __launch_bounds__(256)
k_gemm_bias(const unsigned short* __restrict__ A, int Ttot, int t0, int Tc, int K,
            const unsigned short* __restrict__ W, const float* __restrict__ bias,
            float* __restrict__ Y, int N) {
  const int lane = threadIdx.x & 31, wv = threadIdx.x >> 5;
  const int grp = lane >> 4, lm = lane & 15;
  const int tiles_n = N >> 4;
  const int tile = blockIdx.x * 8 + wv;
  const int tm = tile / tiles_n, tn = tile % tiles_n;
  const int m0 = tm << 4, n0 = tn << 4;

  const int m = m0 + lm;                        // A row this lane reads (M = lane%16)
  const int b = m / Tc, tl = m - b * Tc;
  const unsigned short* arow = A + (unsigned)((b * Ttot + t0 + tl) * K) + (grp << 4);
  const int ncol = n0 + lm;                     // B column this lane reads (N = lane%16)
  const unsigned short* brow = W + (unsigned)(ncol * K) + (grp << 4);

  v8f acc;
  {
    const float bs = bias[ncol];
#pragma unroll
    for (int r = 0; r < 8; ++r) acc[r] = bs;
  }
  for (int kk = 0; kk < K; kk += 32) {
    const v16bf fa = loadA16(arow + kk);
    const v16bf fb = loadB16(brow + kk);
    acc = __builtin_amdgcn_wmma_f32_16x16x32_bf16(false, fa, false, fb, (short)0, acc, false, false);
  }
#pragma unroll
  for (int r = 0; r < 8; ++r)
    Y[(size_t)(m0 + r + (grp << 3)) * N + n0 + lm] = acc[r];
}

// ---------------- sequential LSTM over a time chunk -----------------------------
// grid = 8 blocks: dir = blk>>2 (0=fw,1=bw), p = blk&3 owns hidden slice
// [p*64, p*64+64) i.e. gate columns {g*256 + p*64 .. +64, g=0..3}. Its 256x256
// bf16 Whh slice is LDS-resident (128 KB, loaded once per chunk). Per step the
// 16 KB h vector is exchanged via global hx + a monotonic atomic-counter barrier.
// LDS: shg f32[32][256] 32KB | wlds bf16[256][256] 128KB | shh bf16[32][256] 16KB
#define LSTM_SMEM (32 * 256 * 4 + 256 * 256 * 2 + 32 * 256 * 2)

__global__ void __launch_bounds__(1024)
k_lstm_chunk(const float* __restrict__ Gfw, const float* __restrict__ Gbw,
             const unsigned short* __restrict__ WhhF, const unsigned short* __restrict__ WhhB,
             float* __restrict__ hstate, float* __restrict__ cstate,
             unsigned short* __restrict__ hx, unsigned* __restrict__ cnt,
             unsigned short* __restrict__ out, int Ttot, int t0f, int Tc, int first) {
  extern __shared__ char smem[];
  float*          shg  = (float*)smem;                            // gates fp32
  unsigned short* wlds = (unsigned short*)(smem + 32 * 256 * 4);  // Whh slice
  unsigned short* shh  = wlds + 256 * 256;                        // h bf16 (full)

  const int dir = blockIdx.x >> 2, p = blockIdx.x & 3;
  const float* Gin = dir ? Gbw : Gfw;
  const unsigned short* Whh = dir ? WhhB : WhhF;
  const int coloff = dir * HENC;
  float* hs = hstate + dir * (BB * HENC);
  float* cs = cstate + dir * (BB * HENC);
  unsigned short* hxd = hx + dir * (BB * HENC);
  unsigned* cn = cnt + dir;

  const int tid = threadIdx.x, lane = tid & 31, wv = tid >> 5;
  const int grp = lane >> 4, lm = lane & 15;

  // ---- load Whh slice into LDS: row lc -> global col (lc>>6)*256 + p*64 + (lc&63)
  for (int e = tid; e < 256 * 256; e += 1024) {
    const int lc = e >> 8, k = e & 255;
    const int n = ((lc >> 6) << 8) + (p << 6) + (lc & 63);
    wlds[lc * 256 + k] = Whh[n * 256 + k];
  }
  // ---- init h (full, bf16) and own cell-state slice ----
  float creg[2];
#pragma unroll
  for (int k2 = 0; k2 < 2; ++k2) {
    const int e = tid + (k2 << 10);           // flat over [32][64] own slice
    const int b = e >> 6, ld = e & 63;
    creg[k2] = first ? 0.f : cs[b * HENC + (p << 6) + ld];
  }
#pragma unroll
  for (int k2 = 0; k2 < 8; ++k2) {
    const int f = tid + (k2 << 10);           // flat over [32][256]
    shh[f] = first ? (unsigned short)0 : f2bf(hs[f]);
  }
  __syncthreads();

  // per-wave tile: rows m0 = (wv>>4)*16, gate col block ct = wv&15
  const int m0 = (wv >> 4) << 4;
  const int ct = wv & 15;
  const int ncol = ((ct >> 2) << 8) + (p << 6) + ((ct & 3) << 4) + lm;  // global gate col
  const int lc = (ct << 4) + lm;                                        // local col id

  for (int i = 0; i < Tc; ++i) {
    const int t    = dir ? (Ttot - 1 - t0f - i) : (t0f + i);
    const int tloc = dir ? (Tc - 1 - i) : i;
    // ---- one 16x16 tile per wave, K = 256 from LDS ----
    v8f acc;
#pragma unroll
    for (int r = 0; r < 8; ++r)
      acc[r] = Gin[(unsigned)(((m0 + r + (grp << 3)) * Tc + tloc) * 1024 + ncol)];
    const unsigned short* arow = shh + (m0 + lm) * HENC + (grp << 4);
    const unsigned short* brow = wlds + lc * 256 + (grp << 4);
#pragma unroll
    for (int kk = 0; kk < HENC; kk += 32) {
      acc = __builtin_amdgcn_wmma_f32_16x16x32_bf16(false, loadA16(arow + kk), false,
                                                    loadB16(brow + kk), (short)0, acc, false, false);
    }
#pragma unroll
    for (int r = 0; r < 8; ++r)
      shg[(m0 + r + (grp << 3)) * 256 + lc] = acc[r];
    __syncthreads();
    // ---- cell update: own hidden slice [p*64, p*64+64), 2 elems/thread ----
#pragma unroll
    for (int k2 = 0; k2 < 2; ++k2) {
      const int e = tid + (k2 << 10);
      const int b = e >> 6, ld = e & 63;
      const float gi = shg[b * 256 + ld];
      const float gf = shg[b * 256 + 64 + ld];
      const float gg = shg[b * 256 + 128 + ld];
      const float go = shg[b * 256 + 192 + ld];
      const float c = sigm(gf) * creg[k2] + sigm(gi) * tanhf(gg);
      creg[k2] = c;
      const float h = sigm(go) * tanhf(c);
      const unsigned short hb = f2bf(h);
      out[(size_t)(b * Ttot + t) * (2 * HENC) + coloff + (p << 6) + ld] = hb;
      hxd[b * HENC + (p << 6) + ld] = hb;
      if (i == Tc - 1) { cs[b * HENC + (p << 6) + ld] = c; hs[b * HENC + (p << 6) + ld] = h; }
    }
    __threadfence();          // publish hx slice device-wide
    __syncthreads();
    if (tid == 0) {
      __hip_atomic_fetch_add(cn, 1u, __ATOMIC_RELEASE, __HIP_MEMORY_SCOPE_AGENT);
      const unsigned tgt = 4u * (unsigned)(i + 1);
      while (__hip_atomic_load(cn, __ATOMIC_ACQUIRE, __HIP_MEMORY_SCOPE_AGENT) < tgt)
        __builtin_amdgcn_s_sleep(2);
    }
    __syncthreads();
    // ---- refresh full h from hx ----
#pragma unroll
    for (int k2 = 0; k2 < 8; ++k2) {
      const int f = tid + (k2 << 10);
      shh[f] = hxd[f];
    }
    __syncthreads();
  }
}

// ---------------- decoder: 200 sequential steps, 4 cooperating workgroups -------
// WG p owns d1 hidden slice [p*128,(p+1)*128) = gate cols {g*512 + p*128 .. +128}.
// Per step: all WGs d1 quarter -> publish h1 slice (hx1) -> barrier A -> reload h1.
// WG0 alone: d2 + attention + logits -> publish ctx (ctxg) -> barrier B.
// LDS per WG: g f32[32][512] | sc f32[32][208] | h2f f32[32][128] |
//             yc bf16[32][384] | h1 bf16[32][512] | h2 bf16[32][128]  = 170 KB
#define DEC_SMEM ((32 * 512 + 32 * SENC + 32 * KVD) * 4 + (32 * 384 + 32 * 512 + 32 * KVD) * 2)

__global__ void __launch_bounds__(1024)
k_decoder(const float* __restrict__ key, const float* __restrict__ val,
          const int* __restrict__ x_len, const int* __restrict__ y,
          const float* __restrict__ emb,
          const unsigned short* __restrict__ W1i, const unsigned short* __restrict__ W1h,
          const float* __restrict__ b1,
          const unsigned short* __restrict__ W2i, const unsigned short* __restrict__ W2h,
          const float* __restrict__ b2,
          const float* __restrict__ b_out,
          unsigned short* __restrict__ hx1, unsigned short* __restrict__ ctxg,
          unsigned* __restrict__ cnt, float* __restrict__ out) {
  extern __shared__ char smem[];
  float*          shg = (float*)smem;                        // [32][512] g1 / g2
  float*          sc  = shg + 32 * 512;                      // scores [32][208]
  float*          h2f = sc + 32 * SENC;                      // h2 fp32 [32][128]
  unsigned short* ycb = (unsigned short*)(h2f + 32 * KVD);   // [32][384]
  unsigned short* h1f = ycb + 32 * 384;                      // [32][512] full h1
  unsigned short* h2b = h1f + 32 * 512;                      // [32][128]

  const int p = blockIdx.x;
  const int tid = threadIdx.x, lane = tid & 31, wv = tid >> 5;
  const int grp = lane >> 4, lm = lane & 15;
  const int el = x_len[wv] >> 3;                             // enc_len for batch row wv
  unsigned* cntA = cnt;
  unsigned* cntB = cnt + 1;

  float c1[4], c2r[4];
#pragma unroll
  for (int k = 0; k < 4; ++k) {
    c1[k] = 0.f; c2r[k] = 0.f;
    h2b[tid + (k << 10)] = 0; h2f[tid + (k << 10)] = 0.f;
  }
#pragma unroll
  for (int k = 0; k < 16; ++k) h1f[tid + (k << 10)] = 0;
  __syncthreads();

  auto attend = [&](int t, bool wr) {   // WG0 only; wave wv handles batch row wv
    const int b = wv;
    const float* q = h2f + b * KVD;
    for (int s = lane; s < SENC; s += 32) {
      float e = -3.4028235e38f;
      if (s < el) {
        float a = 0.f;
        const float* kr = key + (unsigned)((b * SENC + s) * KVD);
#pragma unroll 4
        for (int d = 0; d < KVD; ++d) a += q[d] * kr[d];
        e = a * 0.08838834764831845f;   // 1/sqrt(128)
      }
      sc[b * SENC + s] = e;
    }
    float mx = -3.4028235e38f;
    for (int s = lane; s < SENC; s += 32) mx = fmaxf(mx, sc[b * SENC + s]);
#pragma unroll
    for (int o = 16; o; o >>= 1) mx = fmaxf(mx, __shfl_xor(mx, o, 32));
    float sum = 0.f;
    for (int s = lane; s < SENC; s += 32) {
      const float a = __expf(sc[b * SENC + s] - mx);
      sc[b * SENC + s] = a; sum += a;
    }
#pragma unroll
    for (int o = 16; o; o >>= 1) sum += __shfl_xor(sum, o, 32);
    const float inv = 1.f / sum;
    for (int s = lane; s < SENC; s += 32) {
      const float a = sc[b * SENC + s] * inv;
      sc[b * SENC + s] = a;
      if (wr && b == 0) out[PRED + t * SENC + s] = a;
    }
    __syncthreads();
#pragma unroll
    for (int k = 0; k < 4; ++k) {
      const int d = lane + (k << 5);
      float a = 0.f;
      for (int s = 0; s < SENC; ++s)
        a += sc[b * SENC + s] * val[(unsigned)((b * SENC + s) * KVD) + d];
      const unsigned short ab = f2bf(a);
      ycb[b * 384 + 256 + d] = ab;        // ctx lives in yc[:,256:384]
      ctxg[b * KVD + d] = ab;             // and is published for WG1..3
    }
  };

  // ---- ctx0 = attend(0) ----
  if (p == 0) {
    attend(-1, false);
    __threadfence();
    __syncthreads();
    if (tid == 0)
      __hip_atomic_fetch_add(cntB, 1u, __ATOMIC_RELEASE, __HIP_MEMORY_SCOPE_AGENT);
  } else {
    if (tid == 0)
      while (__hip_atomic_load(cntB, __ATOMIC_ACQUIRE, __HIP_MEMORY_SCOPE_AGENT) < 1u)
        __builtin_amdgcn_s_sleep(2);
    __syncthreads();
#pragma unroll
    for (int k = 0; k < 4; ++k) {
      const int e = tid + (k << 10);
      ycb[(e >> 7) * 384 + 256 + (e & 127)] = ctxg[e];
    }
  }
  __syncthreads();

  for (int t = 0; t < LDECS; ++t) {
    // ---- yc[:,0:256] = emb[ins_t] (every WG keeps its own copy) ----
#pragma unroll
    for (int k = 0; k < 8; ++k) {
      const int f = tid + (k << 10);
      const int b = f >> 8, e = f & 255;
      const int idx = (t == 0) ? 0 : y[b * LDECS + t - 1];
      ycb[b * 384 + e] = f2bf(emb[idx * EMBD + e]);
    }
    __syncthreads();
    // ---- d1 quarter: 512 gate cols, 64 tiles, 2 per wave ----
#pragma unroll 1
    for (int q = 0; q < 2; ++q) {
      const int tile = wv + (q << 5);
      const int tm = tile >> 5, ctl = tile & 31;
      const int m0 = tm << 4, bm = m0 + lm;
      const int lc = (ctl << 4) + lm;                       // local col [0,512)
      const int g = lc >> 7, z = lc & 127;
      const int ncol = (g << 9) + (p << 7) + z;             // global gate col [0,2048)
      v8f acc;
      { const float bs = b1[ncol];
#pragma unroll
        for (int r = 0; r < 8; ++r) acc[r] = bs; }
      const unsigned short* a1 = ycb + bm * 384 + (grp << 4);
      const unsigned short* br = W1i + (unsigned)(ncol * 384) + (grp << 4);
#pragma unroll 1
      for (int kk = 0; kk < 384; kk += 32)
        acc = __builtin_amdgcn_wmma_f32_16x16x32_bf16(false, loadA16(a1 + kk), false, loadB16(br + kk), (short)0, acc, false, false);
      const unsigned short* a2 = h1f + bm * 512 + (grp << 4);
      const unsigned short* br2 = W1h + (unsigned)(ncol * 512) + (grp << 4);
#pragma unroll 1
      for (int kk = 0; kk < 512; kk += 32)
        acc = __builtin_amdgcn_wmma_f32_16x16x32_bf16(false, loadA16(a2 + kk), false, loadB16(br2 + kk), (short)0, acc, false, false);
#pragma unroll
      for (int r = 0; r < 8; ++r)
        shg[(m0 + r + (grp << 3)) * 512 + lc] = acc[r];
    }
    __syncthreads();
    // ---- d1 cell: own hidden slice [p*128,(p+1)*128), 4 elems/thread ----
#pragma unroll
    for (int k = 0; k < 4; ++k) {
      const int e = tid + (k << 10);
      const int b = e >> 7, ld = e & 127;
      const float gi = shg[b * 512 + ld],       gf = shg[b * 512 + 128 + ld];
      const float gg = shg[b * 512 + 256 + ld], go = shg[b * 512 + 384 + ld];
      const float c = sigm(gf) * c1[k] + sigm(gi) * tanhf(gg);
      c1[k] = c;
      hx1[b * 512 + (p << 7) + ld] = f2bf(sigm(go) * tanhf(c));
    }
    __threadfence();
    __syncthreads();
    if (tid == 0) {
      __hip_atomic_fetch_add(cntA, 1u, __ATOMIC_RELEASE, __HIP_MEMORY_SCOPE_AGENT);
      const unsigned tgt = 4u * (unsigned)(t + 1);
      while (__hip_atomic_load(cntA, __ATOMIC_ACQUIRE, __HIP_MEMORY_SCOPE_AGENT) < tgt)
        __builtin_amdgcn_s_sleep(2);
    }
    __syncthreads();
    // ---- refresh full h1 ----
#pragma unroll
    for (int k = 0; k < 16; ++k) {
      const int f = tid + (k << 10);
      h1f[f] = hx1[f];
    }
    __syncthreads();

    if (p == 0) {
      // ---- d2: g2 = h1@W2i^T + h2@W2h^T + b2, [32,512], 2 tiles/wave ----
#pragma unroll 1
      for (int q = 0; q < 2; ++q) {
        const int tile = wv + (q << 5);
        const int tm = tile >> 5, tn = tile & 31;
        const int m0 = tm << 4, bm = m0 + lm;
        const int ncol = (tn << 4) + lm;
        v8f acc;
        { const float bs = b2[ncol];
#pragma unroll
          for (int r = 0; r < 8; ++r) acc[r] = bs; }
        const unsigned short* a1 = h1f + bm * 512 + (grp << 4);
        const unsigned short* br = W2i + (unsigned)(ncol * 512) + (grp << 4);
#pragma unroll 1
        for (int kk = 0; kk < 512; kk += 32)
          acc = __builtin_amdgcn_wmma_f32_16x16x32_bf16(false, loadA16(a1 + kk), false, loadB16(br + kk), (short)0, acc, false, false);
        const unsigned short* a2 = h2b + bm * 128 + (grp << 4);
        const unsigned short* br2 = W2h + (unsigned)(ncol * 128) + (grp << 4);
#pragma unroll
        for (int kk = 0; kk < 128; kk += 32)
          acc = __builtin_amdgcn_wmma_f32_16x16x32_bf16(false, loadA16(a2 + kk), false, loadB16(br2 + kk), (short)0, acc, false, false);
#pragma unroll
        for (int r = 0; r < 8; ++r)
          shg[(m0 + r + (grp << 3)) * 512 + ncol] = acc[r];
      }
      __syncthreads();
      // ---- d2 cell ----
#pragma unroll
      for (int k = 0; k < 4; ++k) {
        const int e = tid + (k << 10);
        const int b = e >> 7, hd = e & 127;
        const float gi = shg[b * 512 + hd],       gf = shg[b * 512 + 128 + hd];
        const float gg = shg[b * 512 + 256 + hd], go = shg[b * 512 + 384 + hd];
        const float c = sigm(gf) * c2r[k] + sigm(gi) * tanhf(gg);
        c2r[k] = c;
        const float h = sigm(go) * tanhf(c);
        h2b[e] = f2bf(h);
        h2f[e] = h;
      }
      __syncthreads();
      attend(t, true);
      __syncthreads();
      // ---- logits = [h2|ctx] @ emb^T + b_out (VOCAB=30 -> VALU) ----
      {
        const int b = wv, v = lane;
        if (v < VOC) {
          float s = b_out[v];
          const float* ev = emb + v * EMBD;
#pragma unroll 4
          for (int j = 0; j < KVD; ++j) s += h2f[b * KVD + j] * ev[j];
#pragma unroll 4
          for (int j = 0; j < KVD; ++j) s += bf2f(ycb[b * 384 + 256 + j]) * ev[128 + j];
          out[(size_t)(b * LDECS + t) * VOC + v] = s;
        }
      }
      __threadfence();
      __syncthreads();
      if (tid == 0)
        __hip_atomic_fetch_add(cntB, 1u, __ATOMIC_RELEASE, __HIP_MEMORY_SCOPE_AGENT);
    } else {
      if (tid == 0) {
        const unsigned tgt = (unsigned)(t + 2);   // pre-loop publish + steps 0..t
        while (__hip_atomic_load(cntB, __ATOMIC_ACQUIRE, __HIP_MEMORY_SCOPE_AGENT) < tgt)
          __builtin_amdgcn_s_sleep(2);
      }
      __syncthreads();
#pragma unroll
      for (int k = 0; k < 4; ++k) {
        const int e = tid + (k << 10);
        ycb[(e >> 7) * 384 + 256 + (e & 127)] = ctxg[e];
      }
      __syncthreads();
    }
  }
}

// =======================================================================
extern "C" void kernel_launch(void* const* d_in, const int* in_sizes, int n_in,
                              void* d_out, int out_size, void* d_ws, size_t ws_size,
                              hipStream_t stream) {
  (void)in_sizes; (void)n_in; (void)out_size; (void)ws_size;
  const float* x        = (const float*)d_in[0];
  const int*   x_len    = (const int*)  d_in[1];
  const int*   y        = (const int*)  d_in[2];
  const float* e_fw_Wih = (const float*)d_in[3];
  const float* e_fw_Whh = (const float*)d_in[4];
  const float* e_fw_b   = (const float*)d_in[5];
  const float* e_bw_Wih = (const float*)d_in[6];
  const float* e_bw_Whh = (const float*)d_in[7];
  const float* e_bw_b   = (const float*)d_in[8];
  const float* p_fw_Wih = (const float*)d_in[9];
  const float* p_fw_Whh = (const float*)d_in[10];
  const float* p_fw_b   = (const float*)d_in[11];
  const float* p_bw_Wih = (const float*)d_in[12];
  const float* p_bw_Whh = (const float*)d_in[13];
  const float* p_bw_b   = (const float*)d_in[14];
  const float* Wk       = (const float*)d_in[15];
  const float* bk       = (const float*)d_in[16];
  const float* Wv       = (const float*)d_in[17];
  const float* bv       = (const float*)d_in[18];
  const float* emb      = (const float*)d_in[19];
  const float* d1_Wih   = (const float*)d_in[20];
  const float* d1_Whh   = (const float*)d_in[21];
  const float* d1_b     = (const float*)d_in[22];
  const float* d2_Wih   = (const float*)d_in[23];
  const float* d2_Whh   = (const float*)d_in[24];
  const float* d2_b     = (const float*)d_in[25];
  const float* b_out    = (const float*)d_in[26];

  // ---- workspace layout ----
  char* ws = (char*)d_ws;
  size_t off = 0;
  auto aus = [&](size_t n) { unsigned short* p = (unsigned short*)(ws + off); off += n * 2; return p; };
  unsigned short* bw_e_fw_Wih = aus(1024 * FPAD);          // K-padded 80 -> 96
  unsigned short* bw_e_bw_Wih = aus(1024 * FPAD);
  unsigned short* bw_p_fw_Wih = aus(3 * 1024 * 1024);
  unsigned short* bw_p_bw_Wih = aus(3 * 1024 * 1024);
  unsigned short* bw_e_fw_Whh = aus(1024 * 256);
  unsigned short* bw_e_bw_Whh = aus(1024 * 256);
  unsigned short* bw_p_fw_Whh = aus(3 * 1024 * 256);
  unsigned short* bw_p_bw_Whh = aus(3 * 1024 * 256);
  unsigned short* bw_Wk       = aus(128 * 512);
  unsigned short* bw_Wv       = aus(128 * 512);
  unsigned short* bw_d1_Wih   = aus(2048 * 384);
  unsigned short* bw_d1_Whh   = aus(2048 * 512);
  unsigned short* bw_d2_Wih   = aus(512 * 512);
  unsigned short* bw_d2_Whh   = aus(512 * 128);
  unsigned short* xpad        = aus((size_t)BB * TT * FPAD);   // bf16 padded input
  unsigned short* outA        = aus((size_t)BB * TT * 512);    // layer0/2 output (bf16)
  unsigned short* outB        = aus((size_t)BB * 832 * 512);   // layer1/3 output (bf16)
  unsigned short* hxbuf       = aus(2 * BB * HENC);            // encoder h exchange
  unsigned short* hx1buf      = aus(BB * DDEC);                // decoder h1 exchange
  unsigned short* ctxg        = aus(BB * KVD);                 // decoder ctx exchange
  off = (off + 255) & ~(size_t)255;
  auto af = [&](size_t n) { float* p = (float*)(ws + off); off += n * 4; return p; };
  float* GinF   = af((size_t)BB * TCH * 1024);
  float* GinB   = af((size_t)BB * TCH * 1024);
  float* keyb   = af((size_t)BB * SENC * KVD);
  float* valb   = af((size_t)BB * SENC * KVD);
  float* hstate = af(2 * BB * HENC);
  float* cstate = af(2 * BB * HENC);
  unsigned* cnt = (unsigned*)af(8);       // [0,1] encoder dirs, [2,3] decoder A/B

  // ---- weight conversion / padding ----
  auto cvt = [&](const float* s, unsigned short* d, int n) {
    k_cvt<<<1024, 256, 0, stream>>>(s, d, n);
  };
  k_cvt_pad<<<2048, 256, 0, stream>>>(x, xpad, BB * TT, FIN, FPAD);
  k_cvt_pad<<<512, 256, 0, stream>>>(e_fw_Wih, bw_e_fw_Wih, 1024, FIN, FPAD);
  k_cvt_pad<<<512, 256, 0, stream>>>(e_bw_Wih, bw_e_bw_Wih, 1024, FIN, FPAD);
  cvt(p_fw_Wih, bw_p_fw_Wih, 3 * 1024 * 1024);
  cvt(p_bw_Wih, bw_p_bw_Wih, 3 * 1024 * 1024);
  cvt(e_fw_Whh, bw_e_fw_Whh, 1024 * 256);
  cvt(e_bw_Whh, bw_e_bw_Whh, 1024 * 256);
  cvt(p_fw_Whh, bw_p_fw_Whh, 3 * 1024 * 256);
  cvt(p_bw_Whh, bw_p_bw_Whh, 3 * 1024 * 256);
  cvt(Wk, bw_Wk, 128 * 512);
  cvt(Wv, bw_Wv, 128 * 512);
  cvt(d1_Wih, bw_d1_Wih, 2048 * 384);
  cvt(d1_Whh, bw_d1_Whh, 2048 * 512);
  cvt(d2_Wih, bw_d2_Wih, 512 * 512);
  cvt(d2_Whh, bw_d2_Whh, 512 * 128);

  // ---- encoder base biLSTM (T=1664, K=96 padded) ----
  {
    const int T = TT, nch = T / TCH;              // 16 chunks
    for (int j = 0; j < nch; ++j) {
      const int t0f = j * TCH, t0b = T - (j + 1) * TCH;
      k_gemm_bias<<<1664, 256, 0, stream>>>(xpad, T, t0f, TCH, FPAD, bw_e_fw_Wih, e_fw_b, GinF, 1024);
      k_gemm_bias<<<1664, 256, 0, stream>>>(xpad, T, t0b, TCH, FPAD, bw_e_bw_Wih, e_bw_b, GinB, 1024);
      k_zero_u32<<<1, 32, 0, stream>>>(cnt, 2);
      k_lstm_chunk<<<8, 1024, LSTM_SMEM, stream>>>(GinF, GinB, bw_e_fw_Whh, bw_e_bw_Whh,
                                                   hstate, cstate, hxbuf, cnt, outA, T, t0f, TCH, j == 0);
    }
  }
  // ---- 3 pyramidal biLSTMs (K=1024; frame-pair concat is a free reshape) ----
  {
    const unsigned short* pin[3]  = { outA, outB, outA };
    unsigned short*       pout[3] = { outB, outA, outB };
    const int             Ts[3]   = { 832, 416, 208 };
    for (int L = 0; L < 3; ++L) {
      const int T = Ts[L], nch = T / TCH;
      const unsigned short* WiF = bw_p_fw_Wih + (size_t)L * 1024 * 1024;
      const unsigned short* WiB = bw_p_bw_Wih + (size_t)L * 1024 * 1024;
      const unsigned short* WhF = bw_p_fw_Whh + (size_t)L * 1024 * 256;
      const unsigned short* WhB = bw_p_bw_Whh + (size_t)L * 1024 * 256;
      const float* bF = p_fw_b + L * 1024;
      const float* bB = p_bw_b + L * 1024;
      for (int j = 0; j < nch; ++j) {
        const int t0f = j * TCH, t0b = T - (j + 1) * TCH;
        k_gemm_bias<<<1664, 256, 0, stream>>>(pin[L], T, t0f, TCH, 1024, WiF, bF, GinF, 1024);
        k_gemm_bias<<<1664, 256, 0, stream>>>(pin[L], T, t0b, TCH, 1024, WiB, bB, GinB, 1024);
        k_zero_u32<<<1, 32, 0, stream>>>(cnt, 2);
        k_lstm_chunk<<<8, 1024, LSTM_SMEM, stream>>>(GinF, GinB, WhF, WhB,
                                                     hstate, cstate, hxbuf, cnt, pout[L], T, t0f, TCH, j == 0);
      }
    }
  }
  // ---- key / value projections: [32*208,512] @ [512,128] ----
  k_gemm_bias<<<416, 256, 0, stream>>>(outB, SENC, 0, SENC, 512, bw_Wk, bk, keyb, KVD);
  k_gemm_bias<<<416, 256, 0, stream>>>(outB, SENC, 0, SENC, 512, bw_Wv, bv, valb, KVD);

  // ---- decoder: 4 cooperating workgroups ----
  k_zero_u32<<<1, 32, 0, stream>>>(cnt + 2, 2);
  k_decoder<<<4, 1024, DEC_SMEM, stream>>>(
      keyb, valb, x_len, y, emb,
      bw_d1_Wih, bw_d1_Whh, d1_b, bw_d2_Wih, bw_d2_Whh, d2_b, b_out,
      hx1buf, ctxg, cnt + 2, (float*)d_out);
}
